// CausalSelfAttention_12283606468211
// MI455X (gfx1250) — matrix operand order repaired and checked by reading, compiled, not run on hardware
//
#include <hip/hip_runtime.h>
#include <hip/hip_bf16.h>

typedef __bf16 bf16;
typedef __attribute__((ext_vector_type(16))) __bf16 v16bf;
typedef __attribute__((ext_vector_type(8)))  float  v8f;
typedef __attribute__((ext_vector_type(4)))  unsigned uint32x4;
typedef __attribute__((ext_vector_type(8)))  int      int32x8;
typedef __attribute__((ext_vector_type(4)))  int      int32x4;

#define DEV __device__ __forceinline__

// ---------------------------------------------------------------------------
// Fragment helpers (CDNA5 WMMA 16x16x32 bf16 layouts, wave32)
// A (16x32): lane L -> row M = L%16, h = L/16; vgpr v: K = (v/4)*16 + h*8 + (v%4)*2
// B (32x16): lane L -> N = L%16, h = L/16; element s: K = s<8 ? 8h+s : 16+8h+(s-8)
// C/D (16x16 f32): lane L -> N = L%16; vgpr r -> M = r + 8*(L/16)
// ---------------------------------------------------------------------------

union FragCvt { uint4 q[2]; v16bf v; };

DEV v16bf load_a_frag(const bf16* base, int ld, int row, int kcol, int lane) {
  int h = lane >> 4;
  const bf16* p = base + (size_t)row * ld + kcol + h * 8;
  FragCvt f;
  f.q[0] = *(const uint4*)(p);
  f.q[1] = *(const uint4*)(p + 16);
  return f.v;
}

DEV v16bf load_b_frag(const bf16* base, int ld, int krow, int ncol) {
  v16bf r;
  int hk = krow >> 4;
  for (int s = 0; s < 16; ++s)
    r[s] = base[(size_t)(s < 8 ? 8 * hk + s : 16 + 8 * hk + s - 8) * ld + ncol + (krow & 15)];
  return r;
}

DEV v8f wmma_bf16(v16bf a, v16bf b, v8f c) {
  return __builtin_amdgcn_wmma_f32_16x16x32_bf16(false, a, false, b, (short)0, c,
                                                 false, false);
}

// Low 32 bits of a generic pointer to a __shared__ object == LDS byte offset
// (flat->LDS aperture truncation, ISA 10.2).
DEV unsigned lds_addr_of(const void* p) { return (unsigned)(size_t)p; }

// CDNA5 async DMA: global -> LDS, 16 bytes per lane, tracked by ASYNCcnt.
DEV void async_load_b128(unsigned lds_off, const void* gaddr) {
  asm volatile("global_load_async_to_lds_b128 %0, %1, off"
               :: "v"(lds_off), "v"((unsigned long long)(size_t)gaddr)
               : "memory");
}
DEV void wait_async0() { asm volatile("s_wait_asynccnt 0" ::: "memory"); }

// CDNA5 Tensor Data Mover: 2D tile (tile_d0 x tile_d1 elements of 2 bytes)
// from global (row stride = stride0 elements) into contiguous LDS.
// D# bitfields per ISA 8.3 (group0) / 8.4 (group1); groups 2,3 NULL-equivalent.
// This toolchain exposes the 6-arg builtin form (clang-23 / therock headers).
DEV void tdm_load_2d(unsigned lds_off, const void* gaddr, unsigned tensor_d0,
                     unsigned tensor_d1, unsigned tile_d0, unsigned tile_d1,
                     unsigned stride0) {
  unsigned long long ga = (unsigned long long)(size_t)gaddr;
  uint32x4 g0;
  g0[0] = 1u;                                            // count=1, user mode
  g0[1] = lds_off;                                       // lds_addr [63:32]
  g0[2] = (unsigned)(ga & 0xffffffffu);                  // global_addr lo
  g0[3] = (unsigned)((ga >> 32) & 0x1ffffffu) | (2u << 30);  // addr hi | type=2
  int32x8 g1;
  g1[0] = (int)(1u << 16);                               // data_size=1 (2 bytes)
  g1[1] = (int)((tensor_d0 & 0xffffu) << 16);            // tensor_dim0 lo16
  g1[2] = (int)((tensor_d0 >> 16) | ((tensor_d1 & 0xffffu) << 16));
  g1[3] = (int)((tensor_d1 >> 16) | (tile_d0 << 16));    // tile_dim0
  g1[4] = (int)tile_d1;                                  // tile_dim1, tile_dim2=0
  g1[5] = (int)stride0;                                  // tensor_dim0_stride lo32
  g1[6] = 0;
  g1[7] = 0;
  int32x4 z4 = {0, 0, 0, 0};
  int32x8 z8 = {0, 0, 0, 0, 0, 0, 0, 0};
  __builtin_amdgcn_tensor_load_to_lds(g0, g1, z4, z4, z8, 0);
}

// ---------------------------------------------------------------------------
// fp32 -> bf16 conversion
// ---------------------------------------------------------------------------
__global__ void cvt_f32_bf16(const float* __restrict__ src, bf16* __restrict__ dst,
                             long n) {
  long i = (long)blockIdx.x * blockDim.x + threadIdx.x;
  if (i < n) dst[i] = (bf16)src[i];
}

// ---------------------------------------------------------------------------
// GEMM: out[M,N] = A[M,K] x W[K,N]   (A,W bf16 row-major; out bf16 or f32)
// Workgroup = 4 waves -> 64x64 tile.  W K-chunks (32x64, 4 KB) are moved into
// LDS by the Tensor Data Mover (one descriptor per chunk, issued by wave 0,
// TENSORcnt-tracked), double-buffered so DMA of chunk c+1 overlaps the 4
// WMMAs of chunk c.  B fragments are gathered per element from LDS (128 B per k row).
// ---------------------------------------------------------------------------
__global__ __launch_bounds__(128) void gemm_bf16_wmma(
    const bf16* __restrict__ A, const bf16* __restrict__ W,
    bf16* __restrict__ outb, float* __restrict__ outf, int M, int N, int K) {
  __shared__ bf16 wtile[2][32 * 64];

  int tid = threadIdx.x;
  int wave = tid >> 5, lane = tid & 31;
  int tiles_n = N >> 6;
  int tm = blockIdx.x / tiles_n, tn = blockIdx.x % tiles_n;
  int m0 = (tm << 6) + (wave << 4);
  int n0 = tn << 6;
  unsigned lbase = lds_addr_of(&wtile[0][0]);

  v8f zero = {0.f, 0.f, 0.f, 0.f, 0.f, 0.f, 0.f, 0.f};
  v8f acc[4] = {zero, zero, zero, zero};
  int arow = m0 + (lane & 15);

  if (wave == 0)
    tdm_load_2d(lbase, W + n0, (unsigned)N, (unsigned)K, 64u, 32u, (unsigned)N);

  int nch = K >> 5;
  for (int c = 0; c < nch; ++c) {
    if (wave == 0) __builtin_amdgcn_s_wait_tensorcnt(0);
    __syncthreads();      // tile c visible to all; buffer (c+1)&1 fully consumed
    if (c + 1 < nch && wave == 0)
      tdm_load_2d(lbase + (unsigned)(((c + 1) & 1) * 4096),
                  W + (size_t)((c + 1) << 5) * N + n0, (unsigned)N, (unsigned)K,
                  64u, 32u, (unsigned)N);

    int k0 = c << 5;
    if (k0 + 64 < K)
      __builtin_prefetch(A + (size_t)arow * K + k0 + 64, 0, 0);
    v16bf a = load_a_frag(A, K, arow, k0, lane);
    const bf16* wt = &wtile[c & 1][0];
    v16bf b0, b1, b2, b3;
#pragma unroll
    for (int s = 0; s < 16; ++s) {
      int kr = (s < 8 ? 8 * (lane >> 4) + s : 16 + 8 * (lane >> 4) + s - 8) * 64 + (lane & 15);
      b0[s] = wt[kr];
      b1[s] = wt[kr + 16];
      b2[s] = wt[kr + 32];
      b3[s] = wt[kr + 48];
    }
    acc[0] = wmma_bf16(a, b0, acc[0]);
    acc[1] = wmma_bf16(a, b1, acc[1]);
    acc[2] = wmma_bf16(a, b2, acc[2]);
    acc[3] = wmma_bf16(a, b3, acc[3]);
  }

  int col = lane & 15, h = lane >> 4;
  if (outf) {
#pragma unroll
    for (int nt = 0; nt < 4; ++nt)
#pragma unroll
      for (int r = 0; r < 8; ++r)
        outf[(size_t)(m0 + r + 8 * h) * N + n0 + nt * 16 + col] = acc[nt][r];
  } else {
#pragma unroll
    for (int nt = 0; nt < 4; ++nt)
#pragma unroll
      for (int r = 0; r < 8; ++r)
        outb[(size_t)(m0 + r + 8 * h) * N + n0 + nt * 16 + col] = (bf16)acc[nt][r];
  }
}

// ---------------------------------------------------------------------------
// QKV epilogue: rotary + rmsnorm on q,k ; gate*ve add on v.  One wave per
// (b,t,unit): units 0..15 q-heads, 16..19 k-heads, 20..23 v-heads.
// ---------------------------------------------------------------------------
__global__ void qkv_epilogue(bf16* __restrict__ qb, bf16* __restrict__ kb,
                             bf16* __restrict__ vb, const float* __restrict__ x,
                             const float* __restrict__ ve,
                             const float* __restrict__ cosb,
                             const float* __restrict__ sinb,
                             const float* __restrict__ wgate) {
  const int T = 2048, C = 1024;
  int lane = threadIdx.x & 31;
  long wid = ((long)blockIdx.x * blockDim.x + threadIdx.x) >> 5;
  int u = (int)(wid % 24);
  long bt = wid / 24;
  int t = (int)(bt % T);

  if (u < 20) {
    bf16* buf;
    size_t base;
    if (u < 16) { buf = qb; base = bt * 1024 + (size_t)u * 64; }
    else        { buf = kb; base = bt * 256  + (size_t)(u - 16) * 64; }
    float e1 = (float)buf[base + lane];
    float e2 = (float)buf[base + 32 + lane];
    float c = cosb[(size_t)t * 32 + lane];
    float s = sinb[(size_t)t * 32 + lane];
    float r1 = e1 * c + e2 * s;
    float r2 = -e1 * s + e2 * c;
    float ss = r1 * r1 + r2 * r2;
#pragma unroll
    for (int m = 16; m >= 1; m >>= 1) ss += __shfl_xor(ss, m, 32);
    float rs = rsqrtf(ss * (1.f / 64.f) + 1.1920928955078125e-07f);
    buf[base + lane]      = (bf16)(r1 * rs);
    buf[base + 32 + lane] = (bf16)(r2 * rs);
  } else {
    int kh = u - 20;
    float g = 0.f;
#pragma unroll
    for (int i = 0; i < 8; ++i) g += x[bt * C + i] * wgate[i * 4 + kh];
    g = 2.f / (1.f + __expf(-g));
    size_t base = bt * 256 + (size_t)kh * 64;
    float v1 = (float)vb[base + lane]      + g * ve[base + lane];
    float v2 = (float)vb[base + 32 + lane] + g * ve[base + 32 + lane];
    vb[base + lane]      = (bf16)v1;
    vb[base + 32 + lane] = (bf16)v2;
  }
}

// ---------------------------------------------------------------------------
// Sliding-window GQA flash attention.
// Block = 4 waves = the 4 q-heads sharing one kv-head, same 16-query block:
// K/V 32-key chunks (4 KB each) are staged once per block into LDS with
// global_load_async_to_lds_b128 (double-buffered); every wave reads fragments
// from LDS.  S^T = K x Q^T so softmax stats are per-lane (query = lane%16);
// P round-trips a per-wave LDS slab to become the A fragment of P x V.
// ---------------------------------------------------------------------------
__global__ __launch_bounds__(128) void attn_wmma(
    const bf16* __restrict__ qb, const bf16* __restrict__ kb,
    const bf16* __restrict__ vb, bf16* __restrict__ yb,
    const int* __restrict__ wptr) {
  const int T = 2048;
  const float scale = 0.125f;  // 1/sqrt(64)
  int tid = threadIdx.x;
  int wave = tid >> 5;
  int lane = tid & 31;
  __shared__ bf16 kvtile[2][2][32 * 64];   // [buf][K/V][key][dim]
  __shared__ bf16 plds_all[4][16 * 32];
  bf16* plds = plds_all[wave];
  unsigned kvbase = lds_addr_of(&kvtile[0][0][0]);

  int qblk = blockIdx.x & 127;             // T/16 = 128
  int kh   = (blockIdx.x >> 7) & 3;
  int b    = blockIdx.x >> 9;
  int h    = kh * 4 + wave;                // GQA: 4 q-heads per kv-head
  int q0   = qblk << 4;
  int window = *wptr;

  const bf16* qbase = qb + (size_t)b * T * 1024 + h * 64;
  const bf16* kbase = kb + (size_t)b * T * 256 + kh * 64;
  const bf16* vbase = vb + (size_t)b * T * 256 + kh * 64;

  // Q^T B-fragments (lane%16 = query, elems = head-dim within 32-chunk, table k order)
  v16bf qt[2];
#pragma unroll
  for (int c = 0; c < 2; ++c)
#pragma unroll
    for (int j = 0; j < 16; ++j)
      qt[c][j] = qbase[(size_t)(q0 + (lane & 15)) * 1024 + c * 32 + (j < 8 ? 8 * (lane >> 4) + j : 16 + 8 * (lane >> 4) + j - 8)];

  v8f zero = {0.f, 0.f, 0.f, 0.f, 0.f, 0.f, 0.f, 0.f};
  v8f o[4] = {zero, zero, zero, zero};
  float m = -INFINITY, l = 0.f;

  int qa = q0 + (lane & 15);
  int hh = lane >> 4;
  int q  = lane & 15;

  int k_lo = q0 - window + 1; if (k_lo < 0) k_lo = 0;
  int c_lo = k_lo & ~31;
  int k_hi = q0 + 15;

  // cooperative async stage of one 32-key K+V chunk: 512 x 16B, 4 per thread
  auto stage = [&](int c0, int buf) {
#pragma unroll
    for (int s = 0; s < 4; ++s) {
      int seg = tid + s * 128;
      int which = seg >> 8;                // 0 = K, 1 = V
      int r = (seg >> 3) & 31;
      int cb = seg & 7;
      int krow = c0 + r; if (krow > k_hi) krow = k_hi;   // clamp (masked later)
      const bf16* g = (which ? vbase : kbase) + (size_t)krow * 256 + cb * 8;
      unsigned lofs =
          kvbase + (unsigned)((buf * 2 + which) * 2048 + r * 64 + cb * 8) * 2u;
      async_load_b128(lofs, g);
    }
  };

  int nch = ((k_hi - c_lo) >> 5) + 1;      // block-uniform
  stage(c_lo, 0);
  for (int ci = 0; ci < nch; ++ci) {
    int c0 = c_lo + (ci << 5);
    wait_async0();        // own DMAs done
    __syncthreads();      // everyone's DMAs done; prev buffer fully consumed
    if (ci + 1 < nch) stage(c0 + 32, (ci + 1) & 1);

    const bf16* kt = &kvtile[ci & 1][0][0];
    const bf16* vt = &kvtile[ci & 1][1][0];

    // ---- scores: S^T (two 16-key tiles) from LDS ----
    v8f s[2];
#pragma unroll
    for (int tt = 0; tt < 2; ++tt) {
      int krow = tt * 16 + (lane & 15);
      v16bf ka0 = load_a_frag(kt, 64, krow, 0, lane);
      v16bf ka1 = load_a_frag(kt, 64, krow, 32, lane);
      v8f tmp = wmma_bf16(ka0, qt[0], zero);
      s[tt] = wmma_bf16(ka1, qt[1], tmp);
    }

    // ---- online softmax (per-lane = per-query, merge halves via xor16) ----
    float pv[2][8];
    float tmax = -INFINITY;
#pragma unroll
    for (int tt = 0; tt < 2; ++tt)
#pragma unroll
      for (int r = 0; r < 8; ++r) {
        int ka = c0 + tt * 16 + r + 8 * hh;
        bool valid = (ka <= qa) && (ka > qa - window);
        float val = valid ? s[tt][r] * scale : -INFINITY;
        pv[tt][r] = val;
        tmax = fmaxf(tmax, val);
      }
    tmax = fmaxf(tmax, __shfl_xor(tmax, 16, 32));
    float mn = fmaxf(m, tmax);
    float alpha = (m == -INFINITY) ? 0.f : __expf(m - mn);
    float rsum = 0.f;
#pragma unroll
    for (int tt = 0; tt < 2; ++tt)
#pragma unroll
      for (int r = 0; r < 8; ++r) {
        float p = (pv[tt][r] == -INFINITY) ? 0.f : __expf(pv[tt][r] - mn);
        pv[tt][r] = p;
        rsum += p;
      }
    rsum += __shfl_xor(rsum, 16, 32);
    l = l * alpha + rsum;
    m = mn;

    // ---- P -> per-wave LDS slab (row-major [q][k-offset], bf16) ----
#pragma unroll
    for (int tt = 0; tt < 2; ++tt) {
      union { uint4 u; bf16 e[8]; } pk;
#pragma unroll
      for (int r = 0; r < 8; ++r) pk.e[r] = (bf16)pv[tt][r];
      *(uint4*)&plds[q * 32 + tt * 16 + hh * 8] = pk.u;
    }
    asm volatile("s_wait_dscnt 0" ::: "memory");

    // ---- rescale O by per-row alpha (row = query r+8*hh) ----
#pragma unroll
    for (int r = 0; r < 8; ++r) {
      float ar = __shfl(alpha, r + 8 * hh, 32);
#pragma unroll
      for (int nt = 0; nt < 4; ++nt) o[nt][r] *= ar;
    }

    // ---- O += P x V (V from LDS) ----
    v16bf pa = load_a_frag(plds, 32, q, 0, lane);
#pragma unroll
    for (int nt = 0; nt < 4; ++nt) {
      v16bf vv = load_b_frag(vt, 64, lane, nt * 16);
      o[nt] = wmma_bf16(pa, vv, o[nt]);
    }
  }

  // ---- normalize and store y[b, q, h*64 + d] ----
  int col = lane & 15;
#pragma unroll
  for (int r = 0; r < 8; ++r) {
    float lr = __shfl(l, r + 8 * hh, 32);
    float invl = 1.f / lr;
    size_t row = (size_t)(b * T + q0 + r + 8 * hh) * 1024 + h * 64;
#pragma unroll
    for (int nt = 0; nt < 4; ++nt)
      yb[row + nt * 16 + col] = (bf16)(o[nt][r] * invl);
  }
}

// ---------------------------------------------------------------------------
extern "C" void kernel_launch(void* const* d_in, const int* in_sizes, int n_in,
                              void* d_out, int out_size, void* d_ws, size_t ws_size,
                              hipStream_t stream) {
  const int B = 4, T = 2048, C = 1024, M = B * T;
  const float* x     = (const float*)d_in[0];
  const float* ve    = (const float*)d_in[1];
  const float* cosb  = (const float*)d_in[2];
  const float* sinb  = (const float*)d_in[3];
  const float* wq    = (const float*)d_in[4];
  const float* wk    = (const float*)d_in[5];
  const float* wv    = (const float*)d_in[6];
  const float* wproj = (const float*)d_in[7];
  const float* wgate = (const float*)d_in[8];
  const int*   wind  = (const int*)d_in[9];

  char* ws = (char*)d_ws;
  size_t off = 0;
  auto alloc = [&](size_t bytes) -> void* {
    void* p = ws + off;
    off += (bytes + 255) & ~(size_t)255;
    return p;
  };
  bf16* xb  = (bf16*)alloc((size_t)M * C * 2);
  bf16* wqb = (bf16*)alloc((size_t)C * 1024 * 2);
  bf16* wkb = (bf16*)alloc((size_t)C * 256 * 2);
  bf16* wvb = (bf16*)alloc((size_t)C * 256 * 2);
  bf16* wpb = (bf16*)alloc((size_t)C * 1024 * 2);
  bf16* qbv = (bf16*)alloc((size_t)M * 1024 * 2);
  bf16* kbv = (bf16*)alloc((size_t)M * 256 * 2);
  bf16* vbv = (bf16*)alloc((size_t)M * 256 * 2);
  bf16* yb  = xb;  // x (bf16) is dead after QKV GEMMs; reuse for attn output

  auto cvt = [&](const float* s, bf16* d, long n) {
    cvt_f32_bf16<<<(int)((n + 255) / 256), 256, 0, stream>>>(s, d, n);
  };
  cvt(x, xb, (long)M * C);
  cvt(wq, wqb, (long)C * 1024);
  cvt(wk, wkb, (long)C * 256);
  cvt(wv, wvb, (long)C * 256);
  cvt(wproj, wpb, (long)C * 1024);

  auto gemm = [&](const bf16* A, const bf16* W, bf16* ob, float* of, int Mm,
                  int Nn, int Kk) {
    int tiles = (Mm / 64) * (Nn / 64);
    gemm_bf16_wmma<<<tiles, 128, 0, stream>>>(A, W, ob, of, Mm, Nn, Kk);
  };
  gemm(xb, wqb, qbv, nullptr, M, 1024, 1024);
  gemm(xb, wkb, kbv, nullptr, M, 256, 1024);
  gemm(xb, wvb, vbv, nullptr, M, 256, 1024);

  qkv_epilogue<<<(M * 24) / 8, 256, 0, stream>>>(qbv, kbv, vbv, x, ve, cosb, sinb,
                                                 wgate);

  attn_wmma<<<B * 4 * (T / 16), 128, 0, stream>>>(qbv, kbv, vbv, yb, wind);

  gemm(yb, wpb, nullptr, (float*)d_out, M, 1024, 1024);
}
